// PhaseVortex_88905823027802
// MI455X (gfx1250) — matrix operand order repaired and checked by reading, not compiled, not run on hardware
//
#include <hip/hip_runtime.h>
#include <hip/hip_bf16.h>

typedef _Float16 f16;
typedef __attribute__((ext_vector_type(2)))  _Float16 h2;   // packed complex f16 (re,im)
typedef __attribute__((ext_vector_type(16))) _Float16 v16h;
typedef __attribute__((ext_vector_type(8)))  float    v8f;

#define SEQ   5
#define EDIM  64
#define HDIM  128
#define TB    16                 // batch elements per workgroup
#define ROWS  (TB * SEQ)         // 80 rows of [E]
#define NELEM (ROWS * EDIM)      // 5120 complex elements per workgroup
#define NTHREADS 256
#define PER_TH (NELEM / NTHREADS) // 20 elements owned per thread

// ---- LDS layout (byte offsets into one arena; regions are reused phase to phase)
#define OFF_A    0                       // h/h_op f16 planes, A-swizzled (2 x 5120 halfs);
#define SZ_A     40960                   //   reused as FFN2 f32 output (10240 floats)
#define OFF_Q    (OFF_A + SZ_A)          // Q/K/V packed-complex h2 planes (3 x 5120 h2 = 61440B);
#define SZ_Q     61440                   //   reused as ffn A-swizzled planes (40960B), then h_out f32
#define OFF_W    (OFF_Q + SZ_Q)          // weight f16 planes, B-swizzled (2 x 8192 halfs)
#define SZ_W     32768
#define OFF_ATTN (OFF_W + SZ_W)          // attn weights: 16*5*5 f32
#define SZ_ATTN  (TB * SEQ * SEQ * 4)
#define SMEM_BYTES (OFF_ATTN + SZ_ATTN)  // 136768 B (< 160KB, << 320KB/WGP)

// ---------------- WMMA fragment swizzles (gfx1250 ISA 7.12.2, wave32) ----------------
// Planes are stored already permuted into hardware fragment order, so a lane's
// 16 halfs are one contiguous 32B-aligned run -> 2x ds_load_b128, no repacking.

// A (16x32 f16) tile: lane = (m&15) + 16*h with h=(k>>3)&1 ; element = (k&7) + 8*((k>>4)&1)
__device__ __forceinline__ int a_swz(int m, int k, int ktiles) {
  int mt = m >> 4, mm = m & 15;
  int kt = k >> 5, kk = k & 31;
  int lane = mm + (((kk >> 3) & 1) << 4);
  int e    = (kk & 7) + (((kk >> 4) & 1) << 3);
  return (((mt * ktiles + kt) * 32) + lane) * 16 + e;
}

// B (32x16 f16) tile: lane = (n&15) + 16*h with h=(k>>3)&1 ; element = (k&7) + 8*((k>>4)&1)
__device__ __forceinline__ int b_swz(int k, int n, int ntiles) {
  int kt = k >> 5, kk = k & 31;
  int nt = n >> 4, nn = n & 15;
  int lane = nn + (((kk >> 3) & 1) << 4);
  int e    = (kk & 7) + (((kk >> 4) & 1) << 3);
  return (((kt * ntiles + nt) * 32) + lane) * 16 + e;
}

__device__ __forceinline__ v16h load_a_tile(const f16* __restrict__ plane,
                                            int mt, int kt, int ktiles, int lane) {
  return *(const v16h*)(plane + (((mt * ktiles + kt) * 32) + lane) * 16);
}
__device__ __forceinline__ v16h load_b_tile(const f16* __restrict__ plane,
                                            int kt, int nt, int ntiles, int lane) {
  return *(const v16h*)(plane + (((kt * ntiles + nt) * 32) + lane) * 16);
}

__device__ __forceinline__ v8f wmma_f16(v16h a, v16h b, v8f c) {
  // v_wmma_f32_16x16x32_f16 : D = A(16x32 f16) * B(32x16 f16) + C(16x16 f32)
  return __builtin_amdgcn_wmma_f32_16x16x32_f16(
      false, a, false, b, (short)0, c, false, false);
}

// Complex 16x16 output tile from re/im swizzled planes.
// Four independent accumulators so all 4 WMMAs per k-step are mutually
// independent (no chained WMMA->WMMA RAW inside a k-step; each accumulator's
// self-dependence is 4 WMMAs apart, hiding the XDL pipeline).
// Yr = RR - II ; Yi = RI + IR   (f16 WMMA has no A-negate, combine in VALU).
__device__ __forceinline__ void cgemm_tile(
    const f16* __restrict__ Are, const f16* __restrict__ Aim,
    const f16* __restrict__ Bre, const f16* __restrict__ Bim,
    int mt, int nt, int Kt, int NtB, int lane, v8f& Yr, v8f& Yi) {
  v8f accRR = {};
  v8f accII = {};
  v8f accRI = {};
  v8f accIR = {};
#pragma unroll
  for (int kt = 0; kt < Kt; ++kt) {
    v16h ar = load_a_tile(Are, mt, kt, Kt, lane);
    v16h ai = load_a_tile(Aim, mt, kt, Kt, lane);
    v16h br = load_b_tile(Bre, kt, nt, NtB, lane);
    v16h bi = load_b_tile(Bim, kt, nt, NtB, lane);
    accRR = wmma_f16(ar, br, accRR);
    accII = wmma_f16(ai, bi, accII);
    accRI = wmma_f16(ar, bi, accRI);
    accIR = wmma_f16(ai, br, accIR);
  }
  Yr = accRR - accII;
  Yi = accRI + accIR;
}

// ---------------- fused kernel ----------------

__global__ __launch_bounds__(NTHREADS, 1)
void PhaseVortex_88905823027802_kernel(
    const int*   __restrict__ x,      // [B,S]
    const float* __restrict__ k_dom,  // [B,S,4]
    const float* __restrict__ k_op,   // [B,S,6]
    const float* __restrict__ emb,    // [V,E] c64 (interleaved)
    const float* __restrict__ pos,    // [S,E] c64
    const float* __restrict__ W_dom,  // [E,4]
    const float* __restrict__ W_op,   // [E,6]
    const float* __restrict__ Wq,     // [E,E] c64
    const float* __restrict__ Wk,     // [E,E] c64
    const float* __restrict__ Wv,     // [E,E] c64
    const float* __restrict__ W1,     // [H,E] c64
    const float* __restrict__ W2,     // [E,H] c64
    const float* __restrict__ W_head, // [1,E]
    const float* __restrict__ b_head, // [1]
    float* __restrict__ out_logits,   // [B]
    float* __restrict__ out_h)        // [B,S,E] c64 (interleaved)
{
  __shared__ __align__(64) char smem[SMEM_BYTES];
  f16*   hRe   = (f16*)(smem + OFF_A);          // h / h_op planes (A input, swizzled, Kt=2)
  f16*   hIm   = hRe + NELEM;
  float* yOut  = (float*)(smem + OFF_A);        // reuse: FFN2 complex f32 output
  h2*    qkvC  = (h2*)(smem + OFF_Q);           // 3 packed-complex planes: Q, K, V
  f16*   ffnRe = (f16*)(smem + OFF_Q);          // reuse: relu'd FFN acts (A input, swizzled, Kt=4)
  f16*   ffnIm = ffnRe + ROWS * HDIM;
  float* hOutS = (float*)(smem + OFF_Q);        // reuse: h_out f32 for pooling
  f16*   wRe   = (f16*)(smem + OFF_W);          // weight planes (B input, swizzled)
  f16*   wIm   = wRe + EDIM * HDIM;
  float* attn  = (float*)(smem + OFF_ATTN);     // [TB][SEQ][SEQ]

  const int t    = threadIdx.x;
  const int lane = t & 31;
  const int wave = t >> 5;
  const int b0   = blockIdx.x * TB;

  // warm L2 for later-phase weights while we do the front of the pipeline
  __builtin_prefetch(W1, 0, 1);
  __builtin_prefetch(W2, 0, 1);

  float hr[PER_TH], hi[PER_TH];   // this thread's h elements, carried in VGPRs

  // ---- Phase 0: h = emb[x] + pos, then domain phase rotation -> h_dom
  //      element ownership: flat i = t + 256*j, i = r*64 + e
#pragma unroll
  for (int j = 0; j < PER_TH; ++j) {
    int i = t + NTHREADS * j;
    int r = i >> 6, e = i & 63;
    int bl = r / SEQ, s = r - bl * SEQ;
    int grow = (b0 + bl) * SEQ + s;
    int tok = x[grow];
    float ar = emb[(tok * EDIM + e) * 2]     + pos[(s * EDIM + e) * 2];
    float ai = emb[(tok * EDIM + e) * 2 + 1] + pos[(s * EDIM + e) * 2 + 1];
    float th = 0.f;
#pragma unroll
    for (int d = 0; d < 4; ++d) th += k_dom[grow * 4 + d] * W_dom[e * 4 + d];
    float sn, cs;
    __sincosf(th, &sn, &cs);
    float dr = ar * cs - ai * sn;
    float di = ar * sn + ai * cs;
    hr[j] = dr; hi[j] = di;
    int sa = a_swz(r, e, 2);            // A-plane, K=64 -> 2 k-tiles
    hRe[sa] = (f16)dr; hIm[sa] = (f16)di;
  }
  __syncthreads();

  // ---- Phase 1: Q/K/V = h_dom @ W^T (complex WMMA); weights staged
  //      sequentially through one 32KB B-swizzled region.
  const float* Wqkv[3] = {Wq, Wk, Wv};
#pragma unroll 1
  for (int m = 0; m < 3; ++m) {
    const float* W = Wqkv[m];
    // B[k=e][n=f] = W[f][e]  (einsum 'bse,fe->bsf'), N=64 -> 4 n-tiles
    for (int p = t; p < EDIM * EDIM; p += NTHREADS) {
      int e = p >> 6, f = p & 63;
      int sb = b_swz(e, f, 4);
      wRe[sb] = (f16)W[(f * EDIM + e) * 2];
      wIm[sb] = (f16)W[(f * EDIM + e) * 2 + 1];
    }
    __syncthreads();
    h2* dC = qkvC + m * NELEM;
    for (int tile = wave; tile < 20; tile += 8) {     // 5 M-tiles x 4 N-tiles
      int mt = tile >> 2, nt = tile & 3;
      v8f Yr, Yi;
      cgemm_tile(hRe, hIm, wRe, wIm, mt, nt, 2, 4, lane, Yr, Yi);
      int n = nt * 16 + (lane & 15);
#pragma unroll
      for (int v = 0; v < 8; ++v) {
        int mm = mt * 16 + v + 8 * (lane >> 4);
        dC[mm * EDIM + n] = (h2){(f16)Yr[v], (f16)Yi[v]};  // one b32 store
      }
    }
    __syncthreads();
  }

  // ---- Phase 2: tiny attention: softmax(|Q K^H|/8), 5x5 per batch (VALU)
  const h2* qC = qkvC;
  const h2* kC = qkvC + NELEM;
  const h2* vC = qkvC + 2 * NELEM;
  if (t < ROWS) {                       // one (batch, query-row) per thread
    int bl = t / SEQ, q = t - bl * SEQ;
    float sc[SEQ], mx = -1e30f;
#pragma unroll
    for (int k = 0; k < SEQ; ++k) {
      float dr = 0.f, di = 0.f;
      int qb = (bl * SEQ + q) * EDIM, kb = (bl * SEQ + k) * EDIM;
      for (int e = 0; e < EDIM; ++e) {
        h2 qv = qC[qb + e];
        h2 kv = kC[kb + e];
        float qr = (float)qv[0], qi = (float)qv[1];
        float kr = (float)kv[0], ki = (float)kv[1];
        dr += qr * kr + qi * ki;        // Q * conj(K)
        di += qi * kr - qr * ki;
      }
      float s = sqrtf(dr * dr + di * di) * 0.125f;
      sc[k] = s; mx = fmaxf(mx, s);
    }
    float sum = 0.f;
#pragma unroll
    for (int k = 0; k < SEQ; ++k) { sc[k] = __expf(sc[k] - mx); sum += sc[k]; }
    float inv = 1.f / sum;
#pragma unroll
    for (int k = 0; k < SEQ; ++k) attn[(bl * SEQ + q) * SEQ + k] = sc[k] * inv;
  }
  __syncthreads();

  // ---- Phase 3: h_mid = h_dom + attn@V (registers), then operator rotation
#pragma unroll
  for (int j = 0; j < PER_TH; ++j) {
    int i = t + NTHREADS * j;
    int r = i >> 6, e = i & 63;
    int bl = r / SEQ, s = r - bl * SEQ;
    float ar = hr[j], ai = hi[j];
    const float* aw = attn + (bl * SEQ + s) * SEQ;
#pragma unroll
    for (int k = 0; k < SEQ; ++k) {
      float a = aw[k];
      h2 vv = vC[(bl * SEQ + k) * EDIM + e];
      ar += a * (float)vv[0];
      ai += a * (float)vv[1];
    }
    int grow = (b0 + bl) * SEQ + s;
    float th = 0.f;
#pragma unroll
    for (int o = 0; o < 6; ++o) th += k_op[grow * 6 + o] * W_op[e * 6 + o];
    float sn, cs;
    __sincosf(th, &sn, &cs);
    float orr = ar * cs - ai * sn;
    float oi  = ar * sn + ai * cs;
    hr[j] = orr; hi[j] = oi;
    int sa = a_swz(r, e, 2);            // overwrite h planes with h_op
    hRe[sa] = (f16)orr; hIm[sa] = (f16)oi;
  }
  __syncthreads();

  // ---- Phase 4: z = h_op @ W1^T  ->  split ReLU  ->  ffn A-planes (swizzled, Kt=4)
  for (int p = t; p < EDIM * HDIM; p += NTHREADS) {   // B[k=e][n=h] = W1[h][e], N=128 -> 8 n-tiles
    int e = p >> 7, h = p & 127;
    int sb = b_swz(e, h, 8);
    wRe[sb] = (f16)W1[(h * EDIM + e) * 2];
    wIm[sb] = (f16)W1[(h * EDIM + e) * 2 + 1];
  }
  __syncthreads();
  for (int tile = wave; tile < 40; tile += 8) {       // 5 M-tiles x 8 N-tiles
    int mt = tile >> 3, nt = tile & 7;
    v8f Yr, Yi;
    cgemm_tile(hRe, hIm, wRe, wIm, mt, nt, 2, 8, lane, Yr, Yi);
    int n = nt * 16 + (lane & 15);
#pragma unroll
    for (int v = 0; v < 8; ++v) {
      int mm = mt * 16 + v + 8 * (lane >> 4);
      int sa = a_swz(mm, n, 4);         // becomes A of FFN2 (K=128 -> 4 k-tiles)
      ffnRe[sa] = (f16)fmaxf(Yr[v], 0.f);
      ffnIm[sa] = (f16)fmaxf(Yi[v], 0.f);
    }
  }
  __syncthreads();

  // ---- Phase 5: y = ffn @ W2^T (K = 128), f32 complex output to LDS
  for (int p = t; p < EDIM * HDIM; p += NTHREADS) {   // B[k=h][n=e] = W2[e][h], N=64 -> 4 n-tiles
    int h = p >> 6, e = p & 63;
    int sb = b_swz(h, e, 4);
    wRe[sb] = (f16)W2[(e * HDIM + h) * 2];
    wIm[sb] = (f16)W2[(e * HDIM + h) * 2 + 1];
  }
  __syncthreads();
  for (int tile = wave; tile < 20; tile += 8) {       // 5 M-tiles x 4 N-tiles
    int mt = tile >> 2, nt = tile & 3;
    v8f Yr, Yi;
    cgemm_tile(ffnRe, ffnIm, wRe, wIm, mt, nt, 4, 4, lane, Yr, Yi);
    int n = nt * 16 + (lane & 15);
#pragma unroll
    for (int v = 0; v < 8; ++v) {
      int mm = mt * 16 + v + 8 * (lane >> 4);
      yOut[(mm * EDIM + n) * 2]     = Yr[v];          // adjacent pair ->
      yOut[(mm * EDIM + n) * 2 + 1] = Yi[v];          //   ds_store_b64
    }
  }
  __syncthreads();

  // ---- Phase 6: h_out = h_op + y ; write to HBM and stash f32 for pooling
#pragma unroll
  for (int j = 0; j < PER_TH; ++j) {
    int i = t + NTHREADS * j;
    float orr = hr[j] + yOut[i * 2];
    float oi  = hi[j] + yOut[i * 2 + 1];
    int gi = blockIdx.x * NELEM + i;                  // (b0*S + r)*E + e
    out_h[gi * 2]     = orr;
    out_h[gi * 2 + 1] = oi;
    hOutS[i * 2]     = orr;
    hOutS[i * 2 + 1] = oi;
  }
  __syncthreads();

  // ---- Phase 7: head: sigmoid(W_head . |mean_s h_out| + b)
  if (t < TB) {
    float acc = 0.f;
    for (int e = 0; e < EDIM; ++e) {
      float sr = 0.f, si = 0.f;
#pragma unroll
      for (int s = 0; s < SEQ; ++s) {
        int i = ((t * SEQ + s) * EDIM + e) * 2;
        sr += hOutS[i];
        si += hOutS[i + 1];
      }
      sr *= 0.2f; si *= 0.2f;                         // mean over S=5
      acc += sqrtf(sr * sr + si * si) * W_head[e];
    }
    acc += b_head[0];
    out_logits[b0 + t] = 1.f / (1.f + __expf(-acc));
  }
}

extern "C" void kernel_launch(void* const* d_in, const int* in_sizes, int n_in,
                              void* d_out, int out_size, void* d_ws, size_t ws_size,
                              hipStream_t stream) {
  const int*   x      = (const int*)  d_in[0];
  const float* k_dom  = (const float*)d_in[1];
  const float* k_op   = (const float*)d_in[2];
  const float* emb    = (const float*)d_in[3];   // complex64 interleaved
  const float* pos    = (const float*)d_in[4];
  const float* W_dom  = (const float*)d_in[5];
  const float* W_op   = (const float*)d_in[6];
  const float* Wq     = (const float*)d_in[7];
  const float* Wk     = (const float*)d_in[8];
  const float* Wv     = (const float*)d_in[9];
  const float* W1     = (const float*)d_in[10];
  const float* W2     = (const float*)d_in[11];
  const float* W_head = (const float*)d_in[12];
  const float* b_head = (const float*)d_in[13];

  const int Btot = in_sizes[0] / SEQ;            // 32768
  float* out_logits = (float*)d_out;             // [B]
  float* out_h      = (float*)d_out + Btot;      // [B,S,E] complex64, interleaved

  dim3 grid(Btot / TB), block(NTHREADS);
  PhaseVortex_88905823027802_kernel<<<grid, block, 0, stream>>>(
      x, k_dom, k_op, emb, pos, W_dom, W_op, Wq, Wk, Wv, W1, W2,
      W_head, b_head, out_logits, out_h);
}
